// StructuredLinear_39187281609257
// MI455X (gfx1250) — matrix operand order repaired and checked by reading, not compiled, not run on hardware
//
#include <hip/hip_runtime.h>

typedef __attribute__((ext_vector_type(16))) _Float16 v16h;
typedef __attribute__((ext_vector_type(8)))  _Float16 v8h;
typedef __attribute__((ext_vector_type(8)))  float    v8f;
typedef __attribute__((ext_vector_type(4)))  unsigned int u32x4;
typedef __attribute__((ext_vector_type(8)))  int i32x8;
typedef __attribute__((ext_vector_type(4)))  int i32x4;

#define NN 4096      // layer size
#define RR 4         // displacement rank
#define BATCH 8192   // batch
#define LDS_STRIDE 40  // f16 elements per LDS tile row (80B: 16B-aligned chunks, staggered banks)

// Tensor Data Mover path (clang-23 / therock 6-arg builtin), guarded.
#if defined(__has_builtin)
#  if __has_builtin(__builtin_amdgcn_tensor_load_to_lds) && __has_builtin(__builtin_amdgcn_s_wait_tensorcnt)
#    define USE_TDM 1
#  endif
#endif
#ifndef USE_TDM
#  define USE_TDM 0
#endif

// ---------------------------------------------------------------------------
// WMMA fragment loaders (v_wmma_f32_16x16x32_f16 per-lane layouts, wave32)
// A (16x32, row-major in LDS): lane L holds row L&15;
//   elems 0..7  = K[(L>>4)*8 .. +7],  elems 8..15 = K[(L>>4)*8+16 .. +23]
// B (32x16, stored as Bs[n][k]): lane L holds col L&15;
//   elems 0..7  = K[(L>>4)*8 .. +7],  elems 8..15 = K[(L>>4)*8+16 .. +23]
// ---------------------------------------------------------------------------
__device__ __forceinline__ v16h load_a_frag(const _Float16* base, int lane) {
    const int row  = lane & 15;
    const int half = lane >> 4;
    const v8h lo = *(const v8h*)(base + row * LDS_STRIDE + half * 8);
    const v8h hi = *(const v8h*)(base + row * LDS_STRIDE + half * 8 + 16);
    return __builtin_shufflevector(lo, hi, 0,1,2,3,4,5,6,7,8,9,10,11,12,13,14,15);
}

__device__ __forceinline__ v16h load_b_frag(const _Float16* base, int lane) {
    const int col  = lane & 15;
    const int half = lane >> 4;
    const v8h lo = *(const v8h*)(base + col * LDS_STRIDE + half * 8);
    const v8h hi = *(const v8h*)(base + col * LDS_STRIDE + half * 8 + 16);
    return __builtin_shufflevector(lo, hi, 0,1,2,3,4,5,6,7,8,9,10,11,12,13,14,15);
}

// ---------------------------------------------------------------------------
// Phase 1: W16[i,j] = f16( 0.5 * sum_{r,k} g_r[(i-k)%n] * s(j,k) * h_r[(j-k)%n] )
// GEMM M=4096(i) x N=4096(j) x K=16384(r,k); operand tiles generated on the fly.
// Software-pipelined: gathers for step t+1 fetched to regs before the WMMAs of
// step t, committed to the alternate LDS buffer after. One barrier per step.
// ---------------------------------------------------------------------------
__global__ __launch_bounds__(256)
void krylov_build_w(const float* __restrict__ G, const float* __restrict__ H,
                    _Float16* __restrict__ W16) {
    __shared__ __align__(16) _Float16 As[2][128 * LDS_STRIDE];
    __shared__ __align__(16) _Float16 Bs[2][64 * LDS_STRIDE];

    const int t    = threadIdx.x;
    const int lane = t & 31;
    const int wid  = t >> 5;
    const int wm   = wid >> 1;          // 0..3
    const int wn   = wid & 1;           // 0..1
    const int ibase = blockIdx.x * 128; // i tile
    const int jbase = blockIdx.y * 64;  // j tile

    const v8f vzero = {};
    v8f acc[2][2];
    #pragma unroll
    for (int a = 0; a < 2; ++a)
        #pragma unroll
        for (int b = 0; b < 2; ++b) acc[a][b] = vzero;

    // staging ownership
    const int arow = t >> 1;         // 0..127
    const int akk  = (t & 1) * 16;   // 16 elems each
    const int brow = t >> 2;         // 0..63
    const int bkk  = (t & 3) * 8;    // 8 elems each

    float areg[16], breg[8];

    auto ab_fetch = [&](int kt) {
        const int kglob = kt * 32;
        const int r  = kglob >> 12;        // 32 | 4096 -> step never crosses r
        const int k0 = kglob & (NN - 1);
        const float* g = G + r * NN;       // KA: cyclic shift of g
        const int i = ibase + arow;
        #pragma unroll
        for (int x = 0; x < 16; ++x) {
            const int k = k0 + akk + x;
            areg[x] = g[(i - k) & (NN - 1)];
        }
        const float* h = H + r * NN;       // KB: negacyclic shift of h
        const int j = jbase + brow;
        #pragma unroll
        for (int x = 0; x < 8; ++x) {
            const int k = k0 + bkk + x;
            const float v = h[(j - k) & (NN - 1)];
            breg[x] = (j >= k) ? v : -v;
        }
    };
    auto ab_commit = [&](int buf) {
        #pragma unroll
        for (int x = 0; x < 16; ++x)
            As[buf][arow * LDS_STRIDE + akk + x] = (_Float16)areg[x];
        #pragma unroll
        for (int x = 0; x < 8; ++x)
            Bs[buf][brow * LDS_STRIDE + bkk + x] = (_Float16)breg[x];
    };

    ab_fetch(0);
    ab_commit(0);
    __syncthreads();

    const int KT = (RR * NN) / 32;
    for (int kt = 0; kt < KT; ++kt) {
        const int cur  = kt & 1;
        const bool more = (kt + 1 < KT);
        if (more) ab_fetch(kt + 1);    // global gathers overlap the WMMAs below

        v16h af[2], bf[2];
        #pragma unroll
        for (int fm = 0; fm < 2; ++fm)
            af[fm] = load_a_frag(&As[cur][(wm * 32 + fm * 16) * LDS_STRIDE], lane);
        #pragma unroll
        for (int fn = 0; fn < 2; ++fn)
            bf[fn] = load_b_frag(&Bs[cur][(wn * 32 + fn * 16) * LDS_STRIDE], lane);

        #pragma unroll
        for (int fm = 0; fm < 2; ++fm)
            #pragma unroll
            for (int fn = 0; fn < 2; ++fn)
                acc[fm][fn] = __builtin_amdgcn_wmma_f32_16x16x32_f16(
                    false, af[fm], false, bf[fn], (short)0, acc[fm][fn], false, false);

        if (more) ab_commit(cur ^ 1);
        __syncthreads();
    }

    // store: C/D layout -> row = v + 8*(lane>=16), col = lane&15
    const int col    = lane & 15;
    const int rowoff = (lane >> 4) * 8;
    #pragma unroll
    for (int fm = 0; fm < 2; ++fm)
        #pragma unroll
        for (int fn = 0; fn < 2; ++fn)
            #pragma unroll
            for (int v = 0; v < 8; ++v) {
                const int i = ibase + wm * 32 + fm * 16 + v + rowoff;
                const int j = jbase + wn * 32 + fn * 16 + col;
                W16[(size_t)i * NN + j] = (_Float16)(0.5f * acc[fm][fn][v]);
            }
}

// ---------------------------------------------------------------------------
// Phase 2: out[b,i] = sum_j x[b,j] * W[i,j]   (NT GEMM, W16 is L2-resident)
// M=8192(b) x N=4096(i) x K=4096(j). Block tile 128x128, wave tile 32x64.
// Pipelined double buffer: TDM for tile t+1 issued (and x global loads posted)
// before the WMMAs of tile t; x converted f32->f16 on commit. TDM writes the
// LDS-padded layout directly (16 data DWORDs + 4 pad -> 80B row stride).
// ---------------------------------------------------------------------------
__global__ __launch_bounds__(256)
void gemm_x_wt(const float* __restrict__ X, const _Float16* __restrict__ W16,
               float* __restrict__ Out) {
    __shared__ __align__(16) _Float16 Xs[2][128 * LDS_STRIDE];
    __shared__ __align__(16) _Float16 Ws[2][128 * LDS_STRIDE];

    const int t    = threadIdx.x;
    const int lane = t & 31;
    const int wid  = t >> 5;
    const int wm   = wid >> 1;           // 0..3 -> 32 batch rows each
    const int wn   = wid & 1;            // 0..1 -> 64 i cols each
    const int bbase = blockIdx.y * 128;  // batch tile
    const int ibase = blockIdx.x * 128;  // i tile

    const v8f vzero = {};
    v8f acc[2][4];
    #pragma unroll
    for (int a = 0; a < 2; ++a)
        #pragma unroll
        for (int b = 0; b < 4; ++b) acc[a][b] = vzero;

    const int srow = t >> 1;         // 0..127
    const int skk  = (t & 1) * 16;   // 16 elems each
    const float* xrow = X + (size_t)(bbase + srow) * NN + skk;

    float4 xreg[4];
    auto x_fetch = [&](int kb) {
        const float4* p = (const float4*)(xrow + kb);
        #pragma unroll
        for (int q = 0; q < 4; ++q) xreg[q] = p[q];
    };
    auto x_commit = [&](int buf) {
        const float* xf = (const float*)&xreg[0];
        v8h o0 = {}, o1 = {};
        #pragma unroll
        for (int e = 0; e < 8; ++e) {
            o0[e] = (_Float16)xf[e];
            o1[e] = (_Float16)xf[8 + e];
        }
        *(v8h*)&Xs[buf][srow * LDS_STRIDE + skk]     = o0;
        *(v8h*)&Xs[buf][srow * LDS_STRIDE + skk + 8] = o1;
    };

#if USE_TDM
    const unsigned int ldsWs0 = (unsigned int)(size_t)(&Ws[0][0]);
    const unsigned int ldsWs1 = (unsigned int)(size_t)(&Ws[1][0]);
    const int g1w0 = (1 << 16)      // data_size = 2 bytes
                   | (1 << 20)      // pad_enable
                   | (3 << 22)      // pad_interval: 16 DWORDs (one 64B row)
                   | (3 << 25);     // pad_amount:   4 DWORDs (16B)
    auto tdm_issue = [&](int kb, int buf) {
        if (wid == 0) {  // one wave issues the DMA (TDM ignores EXEC, issues per wave)
            const unsigned long long gaddr =
                (unsigned long long)(size_t)(W16 + (size_t)ibase * NN + kb);
            u32x4 g0;
            g0.x = 1u;                                        // count=1 (valid), user mode
            g0.y = buf ? ldsWs1 : ldsWs0;                     // lds_addr (bytes)
            g0.z = (unsigned int)gaddr;                       // global_addr[31:0]
            g0.w = (unsigned int)(gaddr >> 32) | (2u << 30);  // global_addr[56:32] | type=2
            i32x8 g1;
            g1[0] = g1w0;                    // wg_mask=0 | data_size | padding cfg
            g1[1] = (NN & 0xFFFF) << 16;     // tensor_dim0[15:0] in bits[63:48]
            g1[2] = (NN & 0xFFFF) << 16;     // tensor_dim0 hi=0 | tensor_dim1[15:0]
            g1[3] = (32 << 16);              // tensor_dim1 hi=0 | tile_dim0=32
            g1[4] = 128;                     // tile_dim1=128, tile_dim2=0
            g1[5] = NN;                      // tensor_dim0_stride (elements)
            g1[6] = 0;
            g1[7] = 0;
            const i32x4 gz4 = {0, 0, 0, 0};           // groups 2/3 unused (2D tensor)
            const i32x8 gz8 = {0, 0, 0, 0, 0, 0, 0, 0};
            __builtin_amdgcn_tensor_load_to_lds(g0, g1, gz4, gz4, gz8, 0);
        }
    };
#else
    const _Float16* wrow = W16 + (size_t)(ibase + srow) * NN + skk;
    v8h wreg0, wreg1;
    auto w_fetch = [&](int kb) {
        wreg0 = *(const v8h*)(wrow + kb);
        wreg1 = *(const v8h*)(wrow + kb + 8);
    };
    auto w_commit = [&](int buf) {
        *(v8h*)&Ws[buf][srow * LDS_STRIDE + skk]     = wreg0;
        *(v8h*)&Ws[buf][srow * LDS_STRIDE + skk + 8] = wreg1;
    };
#endif

    // ---- prologue: fill buffer 0 ----
#if USE_TDM
    tdm_issue(0, 0);
#else
    w_fetch(0);
#endif
    x_fetch(0);
    x_commit(0);
#if USE_TDM
    if (wid == 0) __builtin_amdgcn_s_wait_tensorcnt(0);
#else
    w_commit(0);
#endif
    __syncthreads();

    // ---- pipelined main loop ----
    const int KT = NN / 32;
    for (int kt = 0; kt < KT; ++kt) {
        const int cur  = kt & 1;
        const int nxt  = cur ^ 1;
        const bool more = (kt + 1 < KT);
        if (more) {
#if USE_TDM
            tdm_issue((kt + 1) * 32, nxt);   // DMA overlaps the WMMAs below
#else
            w_fetch((kt + 1) * 32);
#endif
            x_fetch((kt + 1) * 32);          // global loads overlap the WMMAs below
        }

        v16h af[2], bf[4];
        #pragma unroll
        for (int fm = 0; fm < 2; ++fm)
            af[fm] = load_a_frag(&Xs[cur][(wm * 32 + fm * 16) * LDS_STRIDE], lane);
        #pragma unroll
        for (int fn = 0; fn < 4; ++fn)
            bf[fn] = load_b_frag(&Ws[cur][(wn * 64 + fn * 16) * LDS_STRIDE], lane);

        #pragma unroll
        for (int fm = 0; fm < 2; ++fm)
            #pragma unroll
            for (int fn = 0; fn < 4; ++fn)
                acc[fm][fn] = __builtin_amdgcn_wmma_f32_16x16x32_f16(
                    false, af[fm], false, bf[fn], (short)0, acc[fm][fn], false, false);

        if (more) {
            x_commit(nxt);
#if USE_TDM
            if (wid == 0) __builtin_amdgcn_s_wait_tensorcnt(0);  // next W tile landed
#else
            w_commit(nxt);
#endif
        }
        __syncthreads();
    }

    const int col    = lane & 15;
    const int rowoff = (lane >> 4) * 8;
    #pragma unroll
    for (int fm = 0; fm < 2; ++fm)
        #pragma unroll
        for (int fn = 0; fn < 4; ++fn)
            #pragma unroll
            for (int v = 0; v < 8; ++v) {
                const int b = bbase + wm * 32 + fm * 16 + v + rowoff;
                const int i = ibase + wn * 64 + fn * 16 + col;
                Out[(size_t)b * NN + i] = acc[fm][fn][v];
            }
}

// ---------------------------------------------------------------------------
extern "C" void kernel_launch(void* const* d_in, const int* in_sizes, int n_in,
                              void* d_out, int out_size, void* d_ws, size_t ws_size,
                              hipStream_t stream) {
    const float* x = (const float*)d_in[0];   // (8192, 4096) f32
    const float* G = (const float*)d_in[1];   // (4, 4096) f32
    const float* H = (const float*)d_in[2];   // (4, 4096) f32
    float*    out  = (float*)d_out;           // (8192, 4096) f32
    _Float16* W16  = (_Float16*)d_ws;         // 4096*4096*2 = 32 MB scratch

    // Phase 1: materialize W (f16) from the rank-4 circulant/skew-circulant factors
    krylov_build_w<<<dim3(NN / 128, NN / 64), 256, 0, stream>>>(G, H, W16);
    // Phase 2: out = x @ W^T
    gemm_x_wt<<<dim3(NN / 128, BATCH / 128), 256, 0, stream>>>(x, W16, out);
}
